// Conv2d_75926431859219
// MI455X (gfx1250) — matrix operand order repaired and checked by reading, not compiled, not run on hardware
//
#include <hip/hip_runtime.h>

// ---- CDNA5 WMMA types -------------------------------------------------------
typedef __attribute__((ext_vector_type(16))) __bf16 v16bf;
typedef __attribute__((ext_vector_type(8)))  float  v8f;

// ---- Problem constants ------------------------------------------------------
#define C_IN   128
#define H_IN   56
#define W_IN   56
#define HW     (H_IN * W_IN)          // 3136
#define K_OUT  256
#define N_IMG  32
#define KRED   (C_IN * 9)             // 1152
#define NSPAT  (N_IMG * HW)           // 100352

// Reduction reordered as k' = tap*C_IN + c (A and B agree -> same GEMM).

// ---- Tile config ------------------------------------------------------------
#define BM     64     // out channels per block
#define BN     256    // spatial columns per block (one per thread)
#define BK     32     // reduction step; 128%32==0 -> tap constant per chunk
#define NCHUNK (KRED / BK)            // 36
#define LDA    40     // padded LDS row stride (ushorts): 64B data + 16B pad
#define LDB    40
// 256 threads = 8 waves arranged 2(M) x 4(N); each wave computes 32x64.

struct F16x2x8 { uint4 lo, hi; };     // 32 bytes == v16bf

// pack two f32 -> packed bf16x2, round-to-nearest (ties up):
// one v_add per input + one v_perm_b32 to merge the two high halves.
__device__ __forceinline__ unsigned f2bf2(float lo, float hi) {
  unsigned a = __builtin_bit_cast(unsigned, lo) + 0x8000u;
  unsigned b = __builtin_bit_cast(unsigned, hi) + 0x8000u;
  // result bytes: [1:0] = a[3:2], [3:2] = b[3:2]
  return __builtin_amdgcn_perm(b, a, 0x07060302u);
}

__global__ __launch_bounds__(256, 2)
void conv2d_implicit_gemm_wmma(const float* __restrict__ x,
                               const float* __restrict__ Kw,
                               const float* __restrict__ bias,
                               float* __restrict__ out) {
  // ping-pong buffers: one barrier per K-chunk
  __shared__ __align__(16) unsigned short ldsA[2][BM * LDA];   // 2 x  5 KB
  __shared__ __align__(16) unsigned short ldsB[2][BN * LDB];   // 2 x 20 KB

  const int t    = threadIdx.x;
  const int lane = t & 31;
  const int wave = t >> 5;
  const int wm   = wave >> 2;                // 0..1  -> M subtile of 32
  const int wn   = wave & 3;                 // 0..3  -> N subtile of 64

  const int blockN0 = blockIdx.x * BN;
  const int blockM0 = blockIdx.y * BM;

  // ---- per-thread spatial decode (thread t owns column j, hoisted) ---------
  const int j  = blockN0 + t;
  const int n  = j / HW;
  const int hw = j - n * HW;
  const int h  = hw / W_IN;
  const int w  = hw - h * W_IN;
  const int xbase = n * (C_IN * HW) + h * W_IN + w;
  const unsigned hOK = (h > 0 ? 1u : 0u) | 2u | (h < H_IN - 1 ? 4u : 0u);
  const unsigned wOK = (w > 0 ? 1u : 0u) | 2u | (w < W_IN - 1 ? 4u : 0u);
  unsigned tapmask = 0;
#pragma unroll
  for (int tp = 0; tp < 9; ++tp)
    tapmask |= (((hOK >> (tp / 3)) & (wOK >> (tp % 3))) & 1u) << tp;

  // ---- weight staging: thread -> (row ml, 8-wide c quarter q) --------------
  const int ml = t >> 2;                      // 0..63
  const int q  = t & 3;                       // 0..3
  const float* __restrict__ arow = Kw + (size_t)(blockM0 + ml) * KRED;
  const int q9 = (q << 3) * 9;

  const float bval = bias[0];

  // ---- raw-register chunk loader (branch-free, clause-friendly) ------------
  auto load_chunk = [&](int k0, float (&ra)[8], float (&rb)[32], unsigned& bm) {
    const int tp  = k0 >> 7;                  // tap, uniform per chunk
    const int c00 = k0 & 127;
    const int dh  = (tp * 11) >> 5;           // tp/3 (exact for 0..8)
    const int dw  = tp - dh * 3;
    // A: OIHW index (c00+q*8+i)*9 + tp, stride 9 -> one clause of 8 b32
    const int abase = c00 * 9 + tp + q9;
#pragma unroll
    for (int i = 0; i < 8; ++i) ra[i] = arow[abase + i * 9];
    // B: always load; invalid tap -> load center tap (in-bounds), mask later
    const unsigned ok = (tapmask >> tp) & 1u;
    bm = 0u - ok;                             // all-ones or zero
    const int tapoff = (dh - 1) * W_IN + (dw - 1);
    const int bbase  = xbase + c00 * HW + (ok ? tapoff : 0);
#pragma unroll
    for (int kl = 0; kl < 32; ++kl) rb[kl] = x[bbase + kl * HW];
  };

  v8f acc[2][4] = {};                         // 2x4 of 16x16 f32 tiles

  float ra[8], rb[32];
  unsigned bm;
  load_chunk(0, ra, rb, bm);                  // prologue

#pragma unroll 1
  for (int i = 0; i < NCHUNK; ++i) {
    // ---- convert chunk i (waits on its loads here), store to lds[i&1] -----
    uint4 qa;
    qa.x = f2bf2(ra[0], ra[1]);  qa.y = f2bf2(ra[2], ra[3]);
    qa.z = f2bf2(ra[4], ra[5]);  qa.w = f2bf2(ra[6], ra[7]);
    uint4 qb[4];
#pragma unroll
    for (int p = 0; p < 4; ++p) {
      qb[p].x = f2bf2(rb[8 * p + 0], rb[8 * p + 1]) & bm;
      qb[p].y = f2bf2(rb[8 * p + 2], rb[8 * p + 3]) & bm;
      qb[p].z = f2bf2(rb[8 * p + 4], rb[8 * p + 5]) & bm;
      qb[p].w = f2bf2(rb[8 * p + 6], rb[8 * p + 7]) & bm;
    }
    unsigned short* lA = ldsA[i & 1];
    unsigned short* lB = ldsB[i & 1];
    *(uint4*)&lA[ml * LDA + (q << 3)] = qa;
    uint4* bd = (uint4*)&lB[t * LDB];
    bd[0] = qb[0]; bd[1] = qb[1]; bd[2] = qb[2]; bd[3] = qb[3];

    // ---- issue chunk i+1 global clause; hides under WMMAs + barrier -------
    if (i + 1 < NCHUNK) load_chunk((i + 1) * BK, ra, rb, bm);

    __syncthreads();   // single barrier per chunk (ping-pong)

    // ---- fragments (contiguous k = 16*(lane/16) + s in A and B) -----------------------------------
    v16bf afrag[2], bfrag[4];
    const int r16  = lane & 15;
    const int aOff = (lane < 16) ? 0 : 16;  // A: lanes hold K 0..15 / 16..31
#pragma unroll
    for (int mt = 0; mt < 2; ++mt) {
      const uint4* p = (const uint4*)&lA[(wm * 32 + mt * 16 + r16) * LDA];
      F16x2x8 f;
      f.lo = p[aOff >> 3];
      f.hi = p[(aOff >> 3) + 1];
      afrag[mt] = __builtin_bit_cast(v16bf, f);
    }
    const int bOff = (lane < 16) ? 0 : 16;  // B: lanes hold K 0..15 / 16..31
#pragma unroll
    for (int nt = 0; nt < 4; ++nt) {
      const uint4* p = (const uint4*)&lB[(wn * 64 + nt * 16 + r16) * LDB + bOff];
      F16x2x8 f;
      f.lo = p[0];
      f.hi = p[1];
      bfrag[nt] = __builtin_bit_cast(v16bf, f);
    }

    // ---- 8 WMMAs per chunk ------------------------------------------------
#pragma unroll
    for (int mt = 0; mt < 2; ++mt)
#pragma unroll
      for (int nt = 0; nt < 4; ++nt)
        acc[mt][nt] = __builtin_amdgcn_wmma_f32_16x16x32_bf16(
            false, afrag[mt], false, bfrag[nt],
            (short)0, acc[mt][nt], false, false);
  }

  // ---- epilogue: C/D layout (VGPR r -> M = r + 8*(lane>=16), N = lane&15) --
#pragma unroll
  for (int mt = 0; mt < 2; ++mt) {
    const int mbase = blockM0 + wm * 32 + mt * 16 + ((lane < 16) ? 0 : 8);
#pragma unroll
    for (int nt = 0; nt < 4; ++nt) {
      const int jo  = blockN0 + wn * 64 + nt * 16 + (lane & 15);
      const int no  = jo / HW;
      const int hwo = jo - no * HW;
      float* op = &out[((size_t)no * K_OUT + mbase) * HW + hwo];
#pragma unroll
      for (int r = 0; r < 8; ++r)
        op[(size_t)r * HW] = acc[mt][nt][r] + bval;   // m = mbase + r
    }
  }
}

// ---- host-side launch -------------------------------------------------------
extern "C" void kernel_launch(void* const* d_in, const int* in_sizes, int n_in,
                              void* d_out, int out_size, void* d_ws, size_t ws_size,
                              hipStream_t stream) {
  (void)in_sizes; (void)n_in; (void)out_size; (void)d_ws; (void)ws_size;
  const float* x    = (const float*)d_in[0];
  const float* Kw   = (const float*)d_in[1];
  const float* bias = (const float*)d_in[2];
  float* out        = (float*)d_out;

  dim3 grid(NSPAT / BN, K_OUT / BM);   // (392, 4) -- exact tiling
  conv2d_implicit_gemm_wmma<<<grid, 256, 0, stream>>>(x, Kw, bias, out);
}
